// MultiHeadSelfAttention_15032385536376
// MI455X (gfx1250) — matrix operand order repaired and checked by reading, not compiled, not run on hardware
//
#include <hip/hip_runtime.h>

#define EMBED 1024
#define HEADS 16
#define HDIM  64
#define SEQ   2048
#define NBATCH 2

typedef __attribute__((ext_vector_type(16))) __bf16 v16bf;
typedef __attribute__((ext_vector_type(8)))  __bf16 v8bf;
typedef __attribute__((ext_vector_type(8)))  float  v8f;
typedef __attribute__((ext_vector_type(4)))  unsigned v4u;
typedef __attribute__((ext_vector_type(8)))  unsigned v8u;

// ---------------------------------------------------------------------------
// Fragment index maps (ISA 7.12.2, wave32):
//  A 16x32 : lane m = lane&15; elems 0..7 <-> K = 8*lh+{0..7}; 8..15 <-> +16
//  B 32x16 : lane n = lane&15; elems 0..7 <-> K = 8*lh+{0..7}; 8..15 <-> +16
//  C/D     : VGPR r <-> M = r + 8*lh, N = lane&15
// ---------------------------------------------------------------------------

__device__ __forceinline__ v16bf cat8(v8bf lo, v8bf hi) {
  return __builtin_shufflevector(lo, hi, 0, 1, 2, 3, 4, 5, 6, 7,
                                 8, 9, 10, 11, 12, 13, 14, 15);
}

__device__ __forceinline__ v8f wmma_bf16(v16bf a, v16bf b, v8f c) {
  return __builtin_amdgcn_wmma_f32_16x16x32_bf16(
      false, a, false, b, (short)0, c, false, false);
}

__device__ __forceinline__ v16bf load_a_frag(const __bf16* rowbase, int k0, int lh) {
  const v8bf lo = *(const v8bf*)(rowbase + k0 + 8 * lh);
  const v8bf hi = *(const v8bf*)(rowbase + k0 + 16 + 8 * lh);
  return cat8(lo, hi);
}

__device__ __forceinline__ v16bf load_b_frag_contig(const __bf16* base, int lh) {
  const v8bf lo = *(const v8bf*)(base + 8 * lh);
  const v8bf hi = *(const v8bf*)(base + 16 + 8 * lh);
  return cat8(lo, hi);
}

// ---------------------------------------------------------------------------
// 16-lane reductions via DPP (replaces ds_bpermute shuffles).
// xor1 = quad_perm[1,0,3,2] (0xB1); xor2 = quad_perm[2,3,0,1] (0x4E);
// then row_half_mirror (0x141) merges quad pairs, row_mirror (0x140) halves.
// ---------------------------------------------------------------------------
template <int CTRL>
__device__ __forceinline__ float dpp_fmax(float x) {
  const int yi = __builtin_amdgcn_update_dpp(
      0, __builtin_bit_cast(int, x), CTRL, 0xF, 0xF, true);
  return fmaxf(x, __builtin_bit_cast(float, yi));
}
template <int CTRL>
__device__ __forceinline__ float dpp_fadd(float x) {
  const int yi = __builtin_amdgcn_update_dpp(
      0, __builtin_bit_cast(int, x), CTRL, 0xF, 0xF, true);
  return x + __builtin_bit_cast(float, yi);
}
__device__ __forceinline__ float red_max16(float x) {
  x = dpp_fmax<0xB1>(x);
  x = dpp_fmax<0x4E>(x);
  x = dpp_fmax<0x141>(x);
  x = dpp_fmax<0x140>(x);
  return x;
}
__device__ __forceinline__ float red_sum16(float x) {
  x = dpp_fadd<0xB1>(x);
  x = dpp_fadd<0x4E>(x);
  x = dpp_fadd<0x141>(x);
  x = dpp_fadd<0x140>(x);
  return x;
}

// ---------------------------------------------------------------------------
// TDM: 2-D tile load Global -> LDS (D# groups 0/1 per ISA ch.8).
// data_size_code: 1 = 2 bytes (bf16), 2 = 4 bytes (f32).
// ---------------------------------------------------------------------------
__device__ __forceinline__ void tdm_load_2d(unsigned lds_addr, const void* gaddr,
                                            unsigned tile_d0, unsigned tile_d1,
                                            unsigned tensor_d0, unsigned tensor_d1,
                                            unsigned stride0_elems,
                                            unsigned data_size_code) {
  const unsigned long long ga = (unsigned long long)(uintptr_t)gaddr;
  v4u g0;
  g0[0] = 1u;                                   // count=1, user descriptor
  g0[1] = lds_addr;                             // LDS byte address
  g0[2] = (unsigned)ga;                         // global_addr[31:0]
  g0[3] = (unsigned)((ga >> 32) & 0x01FFFFFFu)  // global_addr[56:32]
          | 0x80000000u;                        // type = 2 ("image")
  v8u g1;
  g1[0] = data_size_code << 16;
  g1[1] = (tensor_d0 & 0xFFFFu) << 16;          // tensor_dim0[15:0]
  g1[2] = (tensor_d0 >> 16) | ((tensor_d1 & 0xFFFFu) << 16);
  g1[3] = (tensor_d1 >> 16) | (tile_d0 << 16);  // tile_dim0
  g1[4] = tile_d1;                              // tile_dim1 (tile_dim2 = 0)
  g1[5] = stride0_elems;                        // tensor_dim0_stride[31:0]
  g1[6] = 0;
  g1[7] = 0;
  asm volatile("tensor_load_to_lds %0, %1" : : "s"(g0), "s"(g1) : "memory");
}

// ---------------------------------------------------------------------------
// Kernel 0: Wo (1024x1024 f32, row=in) -> WoT (bf16, row=out, col=in)
// ---------------------------------------------------------------------------
__global__ void wo_transpose_kernel(const float* __restrict__ Wo,
                                    __bf16* __restrict__ WoT) {
  const int idx = blockIdx.x * 256 + threadIdx.x;
  const int o = idx >> 10, i = idx & 1023;
  WoT[idx] = (__bf16)Wo[(size_t)i * EMBED + o];
}

// ---------------------------------------------------------------------------
// Kernel 1: per-head projection  X(N,S,H,D) @ W(D,D) + b
//   TP=false -> bf16 [N,H,S,D];  TP=true -> bf16 [N,H,D,S] (for V)
// ---------------------------------------------------------------------------
template <bool TP>
__global__ void proj_kernel(const float* __restrict__ X,
                            const float* __restrict__ W,
                            const float* __restrict__ bias,
                            __bf16* __restrict__ Out) {
  const int lane = threadIdx.x & 31;
  const int lh = lane >> 4, lm = lane & 15;
  int t = blockIdx.x;
  const int dt = t & 3;   t >>= 2;
  const int st = t & 127; t >>= 7;
  const int h  = t & 15;  t >>= 4;
  const int n  = t;
  const int s0 = st * 16, c0 = dt * 16;

  const float* Arow = X + ((size_t)((n * SEQ + s0 + lm) * HEADS) + h) * HDIM;

  v8f acc = {};
#pragma unroll
  for (int kk = 0; kk < HDIM; kk += 32) {
    const v8f c0v = *(const v8f*)(Arow + kk + 8 * lh);
    const v8f c1v = *(const v8f*)(Arow + kk + 16 + 8 * lh);
    v16bf a;
#pragma unroll
    for (int e = 0; e < 8; ++e) {
      a[e]     = (__bf16)c0v[e];
      a[8 + e] = (__bf16)c1v[e];
    }
    v16bf b;
#pragma unroll
    for (int e = 0; e < 16; ++e)
      b[e] = (__bf16)W[(size_t)(kk + (e < 8 ? 8 * lh + e : 16 + 8 * lh + e - 8)) * HDIM + c0 + lm];
    acc = wmma_bf16(a, b, acc);
  }

  const float bval = bias[c0 + lm];
  const size_t head = (size_t)(n * HEADS + h);
#pragma unroll
  for (int r = 0; r < 8; ++r) {
    const int row = r + 8 * lh;
    const __bf16 v = (__bf16)(acc[r] + bval);
    if (!TP)
      Out[(head * SEQ + s0 + row) * HDIM + c0 + lm] = v;
    else
      Out[(head * HDIM + c0 + lm) * SEQ + s0 + row] = v;
  }
}

// ---------------------------------------------------------------------------
// Kernel 2: flash attention. Block = 4 waves, 64 query rows, one (n,h).
//   K (32x64 bf16), V^T (64x32 bf16) and mask (64x32 f32) tiles are
//   double-buffered TDM loads; DMA for kb+32 overlaps compute on kb.
// ---------------------------------------------------------------------------
__global__ void __launch_bounds__(128)
attn_kernel(const __bf16* __restrict__ Qp,
            const __bf16* __restrict__ Kp,
            const __bf16* __restrict__ VpT,
            const float* __restrict__ mask,
            __bf16* __restrict__ Obuf) {
  __shared__ __bf16 KtileS[2][32 * HDIM];  //  8 KB
  __shared__ __bf16 VtS[2][HDIM * 32];     //  8 KB
  __shared__ float  MtS[2][64 * 32];       // 16 KB
  __shared__ __bf16 Plds[4][16 * 32];      //  4 KB

  const int lane = threadIdx.x & 31;
  const int wave = threadIdx.x >> 5;
  const int lh = lane >> 4, lm = lane & 15;
  int t = blockIdx.x;
  const int qb = t & 31; t >>= 5;
  const int h  = t & 15; t >>= 4;
  const int n  = t;
  const int q0 = qb * 64 + wave * 16;

  const size_t headoff = (size_t)(n * HEADS + h) * SEQ * HDIM;
  const __bf16* Qh = Qp + headoff;
  const __bf16* Kh = Kp + headoff;
  const __bf16* Vh = VpT + headoff;                  // [d][s] per head
  const float* maskq = mask + (size_t)(qb * 64) * SEQ;

  // Q: 16x64 as two resident A-fragments (4x global_load_b128)
  const __bf16* qrow = Qh + (size_t)(q0 + lm) * HDIM;
  const v16bf qa0 = load_a_frag(qrow, 0, lh);
  const v16bf qa1 = load_a_frag(qrow, 32, lh);

  float mrow[8], lrow[8];
  v8f o0 = {}, o1 = {}, o2 = {}, o3 = {};
#pragma unroll
  for (int r = 0; r < 8; ++r) { mrow[r] = -1e30f; lrow[r] = 0.0f; }

  const float sc = 0.03125f;  // 1/sqrt(EMBED)

  // Prologue: stage tiles for kb = 0 into buffer 0.
  if (threadIdx.x < 32) {
    tdm_load_2d((unsigned)(uintptr_t)&KtileS[0][0], Kh,
                HDIM, 32, HDIM, SEQ, HDIM, 1);
    tdm_load_2d((unsigned)(uintptr_t)&VtS[0][0], Vh,
                32, HDIM, SEQ, HDIM, SEQ, 1);
    tdm_load_2d((unsigned)(uintptr_t)&MtS[0][0], maskq,
                32, 64, SEQ, 64, SEQ, 2);
  }

  for (int kb = 0; kb < SEQ; kb += 32) {
    const int cur = (kb >> 5) & 1;
    if (threadIdx.x < 32) __builtin_amdgcn_s_wait_tensorcnt(0);
    __syncthreads();
    // Prefetch next tiles into the other buffer (overlaps compute below).
    if (threadIdx.x < 32 && kb + 32 < SEQ) {
      const int nxt = cur ^ 1;
      tdm_load_2d((unsigned)(uintptr_t)&KtileS[nxt][0],
                  Kh + (size_t)(kb + 32) * HDIM, HDIM, 32, HDIM, SEQ, HDIM, 1);
      tdm_load_2d((unsigned)(uintptr_t)&VtS[nxt][0],
                  Vh + kb + 32, 32, HDIM, SEQ, HDIM, SEQ, 1);
      tdm_load_2d((unsigned)(uintptr_t)&MtS[nxt][0],
                  maskq + kb + 32, 32, 64, SEQ, 64, SEQ, 2);
    }
    const __bf16* Kt = &KtileS[cur][0];
    const __bf16* Vt = &VtS[cur][0];
    const float*  Mt = &MtS[cur][0];

    // ---- scores: two 16x16 tiles (keys kb..+15, kb+16..+31) ----
    v8f s0 = {}, s1 = {};
#pragma unroll
    for (int step = 0; step < 2; ++step) {
      const int koff = step * 32;
      const v16bf b0 = load_b_frag_contig(&Kt[lm * HDIM + koff], lh);
      const v16bf b1 = load_b_frag_contig(&Kt[(16 + lm) * HDIM + koff], lh);
      s0 = wmma_bf16(step ? qa1 : qa0, b0, s0);
      s1 = wmma_bf16(step ? qa1 : qa0, b1, s1);
    }

    // ---- multiplicative mask + scale + online softmax (DPP reductions) ----
    float p0[8], p1[8], corr[8];
#pragma unroll
    for (int r = 0; r < 8; ++r) {
      const int rr = wave * 16 + r + 8 * lh;          // row within q-block
      const float e0 = s0[r] * Mt[rr * 32 + lm] * sc;
      const float e1 = s1[r] * Mt[rr * 32 + 16 + lm] * sc;
      const float tmax = red_max16(fmaxf(e0, e1));
      const float mn = fmaxf(mrow[r], tmax);
      corr[r] = __expf(mrow[r] - mn);
      p0[r] = __expf(e0 - mn);
      p1[r] = __expf(e1 - mn);
      const float rs = red_sum16(p0[r] + p1[r]);
      lrow[r] = lrow[r] * corr[r] + rs;
      mrow[r] = mn;
    }
#pragma unroll
    for (int r = 0; r < 8; ++r) {
      o0[r] *= corr[r]; o1[r] *= corr[r];
      o2[r] *= corr[r]; o3[r] *= corr[r];
    }

    // ---- P (16x32) C-layout -> per-wave LDS (DS in-order per wave) ----
    __bf16* Pl = &Plds[wave][0];
#pragma unroll
    for (int r = 0; r < 8; ++r) {
      const int row = r + 8 * lh;
      Pl[row * 32 + lm]      = (__bf16)p0[r];
      Pl[row * 32 + 16 + lm] = (__bf16)p1[r];
    }
    asm volatile("" ::: "memory");
    const v16bf pa = load_a_frag(Pl + lm * 32, 0, lh);
    asm volatile("" ::: "memory");

    // ---- O(16x64) += P(16x32) @ V(32x64); V^T rows contiguous ----
#pragma unroll
    for (int tc = 0; tc < 4; ++tc) {
      const v16bf vb = load_b_frag_contig(&Vt[(tc * 16 + lm) * 32], lh);
      if (tc == 0)      o0 = wmma_bf16(pa, vb, o0);
      else if (tc == 1) o1 = wmma_bf16(pa, vb, o1);
      else if (tc == 2) o2 = wmma_bf16(pa, vb, o2);
      else              o3 = wmma_bf16(pa, vb, o3);
    }
    __syncthreads();  // reads of buffer `cur` done before it is refilled
  }

  // ---- normalize, store bf16 [N, S, EMBED], E = h*64 + d ----
#pragma unroll
  for (int r = 0; r < 8; ++r) {
    const int row = r + 8 * lh;
    const float inv = 1.0f / lrow[r];
    const size_t base = ((size_t)(n * SEQ + q0 + row)) * EMBED + h * HDIM;
    Obuf[base + 0 * 16 + lm] = (__bf16)(o0[r] * inv);
    Obuf[base + 1 * 16 + lm] = (__bf16)(o1[r] * inv);
    Obuf[base + 2 * 16 + lm] = (__bf16)(o2[r] * inv);
    Obuf[base + 3 * 16 + lm] = (__bf16)(o3[r] * inv);
  }
}

// ---------------------------------------------------------------------------
// Kernel 3: output projection  Ob(N*S,1024) @ Wo + bo -> fp32 (uses WoT).
// ---------------------------------------------------------------------------
__global__ void outproj_kernel(const __bf16* __restrict__ A,
                               const __bf16* __restrict__ WoT,
                               const float* __restrict__ bo,
                               float* __restrict__ Out) {
  const int lane = threadIdx.x & 31;
  const int lh = lane >> 4, lm = lane & 15;
  int t = blockIdx.x;
  const int ct = t & 63; t >>= 6;
  const int rt = t;
  const int r0 = rt * 16, c0 = ct * 16;

  const __bf16* arow = A + (size_t)(r0 + lm) * EMBED;
  const __bf16* brow = WoT + (size_t)(c0 + lm) * EMBED;

  v8f acc = {};
  for (int kk = 0; kk < EMBED; kk += 32) {
    const v16bf a = load_a_frag(arow, kk, lh);
    const v16bf b = load_b_frag_contig(brow + kk, lh);
    acc = wmma_bf16(a, b, acc);
  }

  const float bval = bo[c0 + lm];
#pragma unroll
  for (int r = 0; r < 8; ++r) {
    const int row = r0 + r + 8 * lh;
    Out[(size_t)row * EMBED + c0 + lm] = acc[r] + bval;
  }
}

// ---------------------------------------------------------------------------
extern "C" void kernel_launch(void* const* d_in, const int* in_sizes, int n_in,
                              void* d_out, int out_size, void* d_ws, size_t ws_size,
                              hipStream_t stream) {
  (void)in_sizes; (void)n_in; (void)out_size; (void)ws_size;

  const float* values  = (const float*)d_in[0];
  const float* keys    = (const float*)d_in[1];
  const float* queries = (const float*)d_in[2];
  const float* mask    = (const float*)d_in[3];
  const float* Wv = (const float*)d_in[4];
  const float* bv = (const float*)d_in[5];
  const float* Wk = (const float*)d_in[6];
  const float* bk = (const float*)d_in[7];
  const float* Wq = (const float*)d_in[8];
  const float* bq = (const float*)d_in[9];
  const float* Wo = (const float*)d_in[10];
  const float* bo = (const float*)d_in[11];
  float* out = (float*)d_out;

  char* ws = (char*)d_ws;
  const size_t proj_bytes = (size_t)NBATCH * HEADS * SEQ * HDIM * 2;  // 8 MB
  __bf16* Qp  = (__bf16*)(ws + 0 * proj_bytes);
  __bf16* Kp  = (__bf16*)(ws + 1 * proj_bytes);
  __bf16* VpT = (__bf16*)(ws + 2 * proj_bytes);
  __bf16* Ob  = (__bf16*)(ws + 3 * proj_bytes);
  __bf16* WoT = (__bf16*)(ws + 4 * proj_bytes);  // 2 MB

  wo_transpose_kernel<<<EMBED * EMBED / 256, 256, 0, stream>>>(Wo, WoT);

  const int proj_tiles = NBATCH * HEADS * (SEQ / 16) * (HDIM / 16);  // 16384
  proj_kernel<false><<<proj_tiles, 32, 0, stream>>>(queries, Wq, bq, Qp);
  proj_kernel<false><<<proj_tiles, 32, 0, stream>>>(keys,    Wk, bk, Kp);
  proj_kernel<true ><<<proj_tiles, 32, 0, stream>>>(values,  Wv, bv, VpT);

  attn_kernel<<<NBATCH * HEADS * (SEQ / 64), 128, 0, stream>>>(Qp, Kp, VpT, mask, Ob);

  outproj_kernel<<<(NBATCH * SEQ / 16) * (EMBED / 16), 32, 0, stream>>>(Ob, WoT, bo, out);
}
